// MultiDomainPretrainingGNN_75342316307035
// MI455X (gfx1250) — matrix operand order repaired and checked by reading, not compiled, not run on hardware
//
#include <hip/hip_runtime.h>

#define N_EDGES 1000000
#define N_NODES 100000

typedef __attribute__((ext_vector_type(16))) __bf16 bf16x16;
typedef __attribute__((ext_vector_type(8)))  float  f32x8;

union ABReg { bf16x16 v; uint4 q[2]; };

// Native f32->bf16 (RNE) — lowers to v_cvt_pk_bf16_f32 on gfx1250.
static __device__ __forceinline__ unsigned short f2bf(float f) {
  __bf16 h = (__bf16)f;
  return __builtin_bit_cast(unsigned short, h);
}

static __device__ __forceinline__ void wave_lds_sync() {
  // LDS is in-order per wave; this blocks compiler reordering and drains DScnt
  // before cross-lane reads of data written by other lanes of this wave.
  asm volatile("s_wait_dscnt 0x0" ::: "memory");
}

// ws layout (ushort units, each matrix 16B aligned): transposed bf16 weights Wt[N][K]
#define OFF_E1 0
#define OFF_E2 (OFF_E1 + 192 * 128)
#define OFF_E3 (OFF_E2 + 128 * 64)
#define OFF_D1 (OFF_E3 + 64 * 32)
#define OFF_D2 (OFF_D1 + 32 * 64)
#define OFF_D3 (OFF_D2 + 64 * 128)
#define OFF_S1 (OFF_D3 + 128 * 192)

// Transpose+convert W[K][N] f32 -> Wt[N][K] bf16
__global__ void pack_wt(const float* __restrict__ W, unsigned short* __restrict__ Wt,
                        int K, int N) {
  int i = blockIdx.x * blockDim.x + threadIdx.x;
  if (i >= K * N) return;
  int k = i / N, n = i - k * N;
  Wt[(size_t)n * K + k] = f2bf(W[i]);
}

// One 16-row tile: OUT[16][N] = act(A[16][K] @ W + bias)
// src: LDS bf16 activations, row stride 192 ushorts (A-layout read below)
// dst: optional LDS bf16 out (row stride 192); gout: optional global f32 out;
// lout: optional LDS f32 out (row stride 16, N must be 16)
template<int K, int N, bool RELU>
static __device__ __forceinline__ void layer_tile(
    const unsigned short* __restrict__ src,
    unsigned short* __restrict__ dst,
    const unsigned short* __restrict__ wt,
    const float* __restrict__ bias,
    float* __restrict__ gout, int ldo, int grow0,
    float* __restrict__ lout,
    int lane) {
  const int col = lane & 15;   // also the M-row this lane supplies for A
  const int hk  = lane >> 4;
  #pragma unroll 1
  for (int nb = 0; nb < N / 16; ++nb) {
    float bv = bias[nb * 16 + col];
    f32x8 acc = {bv, bv, bv, bv, bv, bv, bv, bv};
    #pragma unroll 2
    for (int kb = 0; kb < K / 32; ++kb) {
      ABReg a, b;
      // A 16x32 bf16: lane<16 -> row=lane, K {kb*32+0..7, +16..23};
      //               lane>=16 -> row=lane-16, K {+8..15, +24..31}
      const unsigned short* ap = src + col * 192 + kb * 32 + hk * 8;
      a.q[0] = *(const uint4*)(ap);
      a.q[1] = *(const uint4*)(ap + 16);
      // B 32x16 bf16 from Wt[N][K]: lane n<16 -> col n, K kb*32+{0..7, 16..23};
      //                             lane n>=16 -> col n-16, K kb*32+{8..15, 24..31}
      const unsigned short* bp = wt + (size_t)(nb * 16 + col) * K + kb * 32 + hk * 8;
      b.q[0] = *(const uint4*)(bp);
      b.q[1] = *(const uint4*)(bp + 16);
      acc = __builtin_amdgcn_wmma_f32_16x16x32_bf16(
          false, a.v, false, b.v, (short)0, acc, false, false);
    }
    // D layout: VGPR j -> row (lane<16 ? j : j+8), col = lane&15
    #pragma unroll
    for (int j = 0; j < 8; ++j) {
      float v = acc[j];
      if (RELU) v = fmaxf(v, 0.0f);
      int row = hk ? (j + 8) : j;
      if (dst)  dst[row * 192 + nb * 16 + col] = f2bf(v);
      if (gout) gout[(size_t)(grow0 + row) * ldo + nb * 16 + col] = v;
      if (lout) lout[row * 16 + nb * 16 + col] = v;
    }
  }
}

__global__ __launch_bounds__(128) void gnn_fused(
    const float* __restrict__ x, const float* __restrict__ ea,
    const int* __restrict__ eidx,
    const unsigned short* __restrict__ ws,
    const float* __restrict__ be1, const float* __restrict__ be2,
    const float* __restrict__ be3, const float* __restrict__ bd1,
    const float* __restrict__ bd2, const float* __restrict__ bd3,
    const float* __restrict__ bs1,
    const float* __restrict__ Ws2, const float* __restrict__ bs2,
    float* __restrict__ out) {
  __shared__ alignas(16) unsigned short s_act[4][2][16 * 192];  // 48 KB
  __shared__ alignas(16) float s_s1[4][16 * 16];                // 4 KB

  const int wv   = threadIdx.x >> 5;
  const int lane = threadIdx.x & 31;
  const int tile = blockIdx.x * 4 + wv;     // 62500 tiles exactly
  const int e0   = tile * 16;

  float* out_enc   = out;
  float* out_rec   = out + (size_t)32  * N_EDGES;
  float* out_trip  = out + (size_t)224 * N_EDGES;
  float* out_score = out + (size_t)416 * N_EDGES;

  unsigned short* buf0 = s_act[wv][0];
  unsigned short* buf1 = s_act[wv][1];

  // ---- gather triplets: [x[src] | edge_attr | x[dst]] -> global f32 + LDS bf16 ----
  #pragma unroll 1
  for (int r = 0; r < 16; ++r) {
    int e = e0 + r;
    int s = eidx[e];
    int d = eidx[N_EDGES + e];
    #pragma unroll
    for (int j = 0; j < 6; ++j) {
      int c = j * 32 + lane;
      float v;
      if (c < 64)       v = x[(size_t)s * 64 + c];
      else if (c < 128) v = ea[(size_t)e * 64 + (c - 64)];
      else              v = x[(size_t)d * 64 + (c - 128)];
      out_trip[(size_t)e * 192 + c] = v;
      buf0[r * 192 + c] = f2bf(v);
    }
  }
  wave_lds_sync();

  // encoder: 192 -> 128 -> 64 -> 32
  layer_tile<192, 128, true >(buf0, buf1, ws + OFF_E1, be1, nullptr, 0, 0, nullptr, lane);
  wave_lds_sync();
  layer_tile<128,  64, true >(buf1, buf0, ws + OFF_E2, be2, nullptr, 0, 0, nullptr, lane);
  wave_lds_sync();
  layer_tile< 64,  32, false>(buf0, buf1, ws + OFF_E3, be3, out_enc, 32, e0, nullptr, lane);
  wave_lds_sync();

  // scorer hidden: 32 -> 16 (keep f32 in LDS)
  layer_tile< 32,  16, true >(buf1, nullptr, ws + OFF_S1, bs1, nullptr, 0, 0, s_s1[wv], lane);

  // decoder: 32 -> 64 -> 128 -> 192
  layer_tile< 32,  64, true >(buf1, buf0, ws + OFF_D1, bd1, nullptr, 0, 0, nullptr, lane);
  wave_lds_sync();
  layer_tile< 64, 128, true >(buf0, buf1, ws + OFF_D2, bd2, nullptr, 0, 0, nullptr, lane);
  wave_lds_sync();
  layer_tile<128, 192, false>(buf1, nullptr, ws + OFF_D3, bd3, out_rec, 192, e0, nullptr, lane);

  // scorer output: 16 -> 1, sigmoid
  wave_lds_sync();
  if (lane < 16) {
    float a = bs2[0];
    const float* sr = s_s1[wv] + lane * 16;
    #pragma unroll
    for (int n = 0; n < 16; ++n) a += sr[n] * Ws2[n];
    out_score[e0 + lane] = 1.0f / (1.0f + __expf(-a));
  }
}

extern "C" void kernel_launch(void* const* d_in, const int* in_sizes, int n_in,
                              void* d_out, int out_size, void* d_ws, size_t ws_size,
                              hipStream_t stream) {
  const float* x   = (const float*)d_in[0];
  const float* ea  = (const float*)d_in[1];
  const int*   ei  = (const int*)d_in[2];
  const float* We1 = (const float*)d_in[3];  const float* be1 = (const float*)d_in[4];
  const float* We2 = (const float*)d_in[5];  const float* be2 = (const float*)d_in[6];
  const float* We3 = (const float*)d_in[7];  const float* be3 = (const float*)d_in[8];
  const float* Wd1 = (const float*)d_in[9];  const float* bd1 = (const float*)d_in[10];
  const float* Wd2 = (const float*)d_in[11]; const float* bd2 = (const float*)d_in[12];
  const float* Wd3 = (const float*)d_in[13]; const float* bd3 = (const float*)d_in[14];
  const float* Ws1 = (const float*)d_in[15]; const float* bs1 = (const float*)d_in[16];
  const float* Ws2 = (const float*)d_in[17]; const float* bs2 = (const float*)d_in[18];
  unsigned short* ws = (unsigned short*)d_ws;

  struct P { const float* W; int off, K, N; };
  const P packs[7] = {
      {We1, OFF_E1, 192, 128}, {We2, OFF_E2, 128, 64}, {We3, OFF_E3, 64, 32},
      {Wd1, OFF_D1, 32, 64},   {Wd2, OFF_D2, 64, 128}, {Wd3, OFF_D3, 128, 192},
      {Ws1, OFF_S1, 32, 16}};
  for (int i = 0; i < 7; ++i) {
    int n = packs[i].K * packs[i].N;
    pack_wt<<<(n + 255) / 256, 256, 0, stream>>>(packs[i].W, ws + packs[i].off,
                                                 packs[i].K, packs[i].N);
  }

  gnn_fused<<<N_EDGES / 64, 128, 0, stream>>>(
      x, ea, ei, ws, be1, be2, be3, bd1, bd2, bd3, bs1, Ws2, bs2, (float*)d_out);
}
